// SpatialHRVQTSSM_16621523435914
// MI455X (gfx1250) — matrix operand order repaired and checked by reading, not compiled, not run on hardware
//
#include <hip/hip_runtime.h>
#include <hip/hip_bf16.h>

// ---------------------------------------------------------------------------
// SpatialHRVQTSSM forward, fused bf16-WMMA pipeline for gfx1250 (MI455X)
//   proj GEMM -> concat(pos_emb) -> mlp1 GEMM + SiLU -> mlp2 GEMM (logits)
//   -> per-row 3-level argmax + embedding gather -> agg GEMM (stoch)
// GEMM staging uses CDNA5 async global->LDS DMA (ASYNCcnt), so no VGPRs are
// held across the WMMA body (the register-staged variant spilled to scratch).
// ---------------------------------------------------------------------------

typedef __attribute__((ext_vector_type(16))) __bf16 v16bf;
typedef __attribute__((ext_vector_type(8)))  float  v8f;
typedef __attribute__((ext_vector_type(4)))  int    v4i_t;

#define BM 128
#define BN 128
#define BK 32

#if __has_builtin(__builtin_amdgcn_global_load_async_to_lds_b128) && \
    __has_builtin(__builtin_amdgcn_s_wait_asynccnt)
#define HAVE_ASYNC_LDS 1
#else
#define HAVE_ASYNC_LDS 0
#endif

static __device__ __forceinline__ unsigned short f2bf(float f) {
  unsigned u = __float_as_uint(f);
  u += 0x7fffu + ((u >> 16) & 1u);          // round-to-nearest-even
  return (unsigned short)(u >> 16);
}

union FragH { uint4 u[2]; v16bf v; };

#if HAVE_ASYNC_LDS
typedef __attribute__((address_space(1))) v4i_t gv4i;
typedef __attribute__((address_space(3))) v4i_t lv4i;
static __device__ __forceinline__ void async_cp16(const void* g, void* l) {
  __builtin_amdgcn_global_load_async_to_lds_b128(
      (gv4i*)(unsigned long long)g,
      (lv4i*)(unsigned)(unsigned long long)l,   // LDS offset = low 32 bits
      /*offset=*/0, /*cpol=*/0);
}
#endif

// --------------------------- generic bf16 GEMM -----------------------------
// C(M,N) = A(M,K) * Bt(N,K)^T + bias.  Block tile 128x128, 8 waves (wave32),
// each wave computes 32x64 via 2x4 grid of 16x16 WMMA accumulators.
// MODE 0: proj   -> bf16 scatter into x[(row*16 + c/128)*160 + c%128]
// MODE 1: mlp1   -> SiLU, bf16 into h[row*512 + c]
// MODE 2: mlp2   -> fp32 logits (NT) with level split
// MODE 3: agg    -> fp32 out[row*1024 + c]
template <int MODE>
__global__ __launch_bounds__(256)
void gemm_bf16_wmma(const unsigned short* __restrict__ A, int lda,
                    const unsigned short* __restrict__ Bt,  // (N x K) bf16
                    const float* __restrict__ bias,
                    float* __restrict__ outF,
                    unsigned short* __restrict__ outH,
                    int K) {
  __shared__ __align__(16) unsigned short lA[2 * BM * BK];   // 2 x 128x32
  __shared__ __align__(16) unsigned short lB[2 * BN * BK];   // 2 x 128x32 (n-major)

  const int tid  = threadIdx.x;
  const int lane = tid & 31;
  const int wv   = tid >> 5;
  const int wm   = wv & 3;        // wave row-tile: rows [wm*32, +32)
  const int wn   = wv >> 2;       // wave col-group: cols [wn*64, +64)
  const int half = lane >> 4;
  const int l16  = lane & 15;

  const long long bm = (long long)blockIdx.y * BM;
  const int       bn = blockIdx.x * BN;

  v8f c[2][4];
#pragma unroll
  for (int mi = 0; mi < 2; ++mi)
#pragma unroll
    for (int ni = 0; ni < 4; ++ni) c[mi][ni] = (v8f){};

  // preload bias for this thread's 4 column slots
  float biasr[4];
#pragma unroll
  for (int ni = 0; ni < 4; ++ni) biasr[ni] = bias[bn + wn * 64 + ni * 16 + l16];

  // staging: 2x 16B chunks per thread per tile; element e -> row e>>2, (e&3)*8
  int srow[2], skq[2];
#pragma unroll
  for (int i = 0; i < 2; ++i) {
    const int e = tid + i * 256;
    srow[i] = e >> 2;
    skq[i]  = (e & 3) * 8;
  }

#if HAVE_ASYNC_LDS
  // ---------- async global->LDS double-buffered pipeline ----------
  // prologue: tile 0 -> buffer 0 (4 async ops per thread)
#pragma unroll
  for (int i = 0; i < 2; ++i) {
    async_cp16(&A[(bm + srow[i]) * (long long)lda + skq[i]],
               &lA[0 * (BM * BK) + srow[i] * BK + skq[i]]);
    async_cp16(&Bt[(long long)(bn + srow[i]) * K + skq[i]],
               &lB[0 * (BN * BK) + srow[i] * BK + skq[i]]);
  }
  int buf = 0;
  for (int kk = 0; kk < K; kk += BK) {
    if (kk + BK < K) {   // issue next tile's DMA into the other buffer
      const int nb = buf ^ 1;
#pragma unroll
      for (int i = 0; i < 2; ++i) {
        async_cp16(&A[(bm + srow[i]) * (long long)lda + kk + BK + skq[i]],
                   &lA[nb * (BM * BK) + srow[i] * BK + skq[i]]);
        async_cp16(&Bt[(long long)(bn + srow[i]) * K + kk + BK + skq[i]],
                   &lB[nb * (BN * BK) + srow[i] * BK + skq[i]]);
      }
      __builtin_amdgcn_s_wait_asynccnt(4);  // current buf's 4 ops done
    } else {
      __builtin_amdgcn_s_wait_asynccnt(0);
    }
    __syncthreads();     // buf visible to all waves

    FragH a[2], b[4];
#pragma unroll
    for (int mi = 0; mi < 2; ++mi) {
      const unsigned short* ap =
          &lA[buf * (BM * BK) + (wm * 32 + mi * 16 + l16) * BK];
      a[mi].u[0] = *(const uint4*)&ap[half * 8];
      a[mi].u[1] = *(const uint4*)&ap[16 + half * 8];
    }
#pragma unroll
    for (int ni = 0; ni < 4; ++ni) {
      const unsigned short* bp =
          &lB[buf * (BN * BK) + (wn * 64 + ni * 16 + l16) * BK];
      b[ni].u[0] = *(const uint4*)&bp[half * 8];
      b[ni].u[1] = *(const uint4*)&bp[16 + half * 8];
    }
#pragma unroll
    for (int mi = 0; mi < 2; ++mi)
#pragma unroll
      for (int ni = 0; ni < 4; ++ni)
        c[mi][ni] = __builtin_amdgcn_wmma_f32_16x16x32_bf16(
            false, a[mi].v, false, b[ni].v, (short)0, c[mi][ni], false, false);

    __syncthreads();     // all reads of buf done before it is DMA'd again
    buf ^= 1;
  }
#else
  // ---------- fallback: transient-register staging, single buffer ----------
  for (int kk = 0; kk < K; kk += BK) {
    __syncthreads();
#pragma unroll
    for (int i = 0; i < 2; ++i) {
      uint4 ra = *(const uint4*)&A[(bm + srow[i]) * (long long)lda + kk + skq[i]];
      uint4 rb = *(const uint4*)&Bt[(long long)(bn + srow[i]) * K + kk + skq[i]];
      *(uint4*)&lA[srow[i] * BK + skq[i]] = ra;
      *(uint4*)&lB[srow[i] * BK + skq[i]] = rb;
    }
    if (kk + BK < K)
      __builtin_prefetch(&A[(bm + srow[0]) * (long long)lda + kk + BK], 0, 1);
    __syncthreads();

    FragH a[2], b[4];
#pragma unroll
    for (int mi = 0; mi < 2; ++mi) {
      const unsigned short* ap = &lA[(wm * 32 + mi * 16 + l16) * BK];
      a[mi].u[0] = *(const uint4*)&ap[half * 8];
      a[mi].u[1] = *(const uint4*)&ap[16 + half * 8];
    }
#pragma unroll
    for (int ni = 0; ni < 4; ++ni) {
      const unsigned short* bp = &lB[(wn * 64 + ni * 16 + l16) * BK];
      b[ni].u[0] = *(const uint4*)&bp[half * 8];
      b[ni].u[1] = *(const uint4*)&bp[16 + half * 8];
    }
#pragma unroll
    for (int mi = 0; mi < 2; ++mi)
#pragma unroll
      for (int ni = 0; ni < 4; ++ni)
        c[mi][ni] = __builtin_amdgcn_wmma_f32_16x16x32_bf16(
            false, a[mi].v, false, b[ni].v, (short)0, c[mi][ni], false, false);
  }
#endif

  // epilogue: acc VGPR r -> M = (half ? r+8 : r), N = l16
#pragma unroll
  for (int mi = 0; mi < 2; ++mi) {
#pragma unroll
    for (int ni = 0; ni < 4; ++ni) {
      const int col = bn + wn * 64 + ni * 16 + l16;
#pragma unroll
      for (int r = 0; r < 8; ++r) {
        const long long row = bm + wm * 32 + mi * 16 + (half ? r + 8 : r);
        float v = c[mi][ni][r] + biasr[ni];
        if (MODE == 0) {
          outH[(row * 16 + (col >> 7)) * 160 + (col & 127)] = f2bf(v);
        } else if (MODE == 1) {
          v = v / (1.f + __expf(-v));
          outH[row * 512 + col] = f2bf(v);
        } else if (MODE == 2) {
          const long long LVL = 131072ll * 512ll;
          __builtin_nontemporal_store(
              v, &outF[(long long)(col >> 9) * LVL + row * 512 + (col & 511)]);
        } else {
          outF[row * 1024 + col] = v;
        }
      }
    }
  }
}

// --------------------------- helper kernels --------------------------------
__global__ void f32_to_bf16_k(const float* __restrict__ in,
                              unsigned short* __restrict__ out, int n) {
  int i = blockIdx.x * 256 + threadIdx.x;
  if (i < n) out[i] = f2bf(in[i]);
}

// transpose-convert: in (K x N) fp32 -> out (N x K) bf16, 32x32 LDS tiles
__global__ __launch_bounds__(256)
void f32_to_bf16_T_k(const float* __restrict__ in,
                     unsigned short* __restrict__ out, int K, int N) {
  __shared__ float t[32][33];
  const int bk = blockIdx.x * 32, bn = blockIdx.y * 32;
  const int tx = threadIdx.x & 31, ty = threadIdx.x >> 5;   // 32 x 8
#pragma unroll
  for (int i = 0; i < 32; i += 8)
    t[ty + i][tx] = in[(long long)(bk + ty + i) * N + bn + tx];
  __syncthreads();
#pragma unroll
  for (int i = 0; i < 32; i += 8)
    out[(long long)(bn + ty + i) * K + bk + tx] = f2bf(t[tx][ty + i]);
}

// fill pos_emb part of x: x[row*160 + 128 + j] = pos_emb[(row&15)*32 + j]
__global__ void fill_posemb_k(const float* __restrict__ pe,
                              unsigned short* __restrict__ x) {
  int i = blockIdx.x * 256 + threadIdx.x;   // 131072*32 threads
  int row = i >> 5, j = i & 31, p = row & 15;
  x[(long long)row * 160 + 128 + j] = f2bf(pe[p * 32 + j]);
}

// one wave (32 lanes) per row: 3x argmax over 512 + residual gather into z_q
__global__ __launch_bounds__(256)
void argmax_gather_k(const float* __restrict__ logits,  // 3 x (131072 x 512)
                     float* __restrict__ idx_out,       // 3 x 131072
                     const float* __restrict__ e0,
                     const float* __restrict__ e1,
                     const float* __restrict__ e2,
                     unsigned short* __restrict__ zq)    // 131072 x 256 bf16
{
  const int lane = threadIdx.x & 31;
  const long long row = (long long)blockIdx.x * 8 + (threadIdx.x >> 5);
  const float* embs[3] = {e0, e1, e2};
  const long long LVL = 131072ll * 512ll;

  float acc[8];
#pragma unroll
  for (int i = 0; i < 8; ++i) acc[i] = 0.f;

  for (int l = 0; l < 3; ++l) {
    const float* lg = logits + (long long)l * LVL + row * 512;
    float best = -3.402823e38f;
    int bi = 0;
#pragma unroll
    for (int i = 0; i < 16; ++i) {               // ascending idx -> first max
      const int ci = lane + i * 32;
      const float v = lg[ci];
      if (v > best) { best = v; bi = ci; }
    }
#pragma unroll
    for (int off = 16; off > 0; off >>= 1) {     // wave32 reduce, low-idx tie
      float ov = __shfl_xor(best, off, 32);
      int   oi = __shfl_xor(bi, off, 32);
      if (ov > best || (ov == best && oi < bi)) { best = ov; bi = oi; }
    }
    if (lane == 0) idx_out[(long long)l * 131072 + row] = (float)bi;
    const float* e = embs[l] + (long long)bi * 256;
#pragma unroll
    for (int i = 0; i < 8; ++i) acc[i] += e[lane + i * 32];
  }
#pragma unroll
  for (int i = 0; i < 8; ++i)
    zq[row * 256 + lane + i * 32] = f2bf(acc[i]);
}

// ------------------------------- launch ------------------------------------
extern "C" void kernel_launch(void* const* d_in, const int* in_sizes, int n_in,
                              void* d_out, int out_size, void* d_ws, size_t ws_size,
                              hipStream_t stream) {
  const float* deter   = (const float*)d_in[0];
  const float* proj_w  = (const float*)d_in[1];
  const float* proj_b  = (const float*)d_in[2];
  const float* pos_emb = (const float*)d_in[3];
  const float* mlp_w1  = (const float*)d_in[4];
  const float* mlp_b1  = (const float*)d_in[5];
  const float* mlp_w2  = (const float*)d_in[6];
  const float* mlp_b2  = (const float*)d_in[7];
  const float* emb_l0  = (const float*)d_in[8];
  const float* emb_l1  = (const float*)d_in[9];
  const float* emb_l2  = (const float*)d_in[10];
  const float* agg_w   = (const float*)d_in[11];
  const float* agg_b   = (const float*)d_in[12];
  float* out = (float*)d_out;

  char* ws = (char*)d_ws;  // all offsets 256B-aligned
  unsigned short* deterH = (unsigned short*)(ws);              // 8192*1024
  unsigned short* projwT = (unsigned short*)(ws + 16777216);   // 2048*1024
  unsigned short* w1T    = (unsigned short*)(ws + 20971520);   // 512*160
  unsigned short* w2T    = (unsigned short*)(ws + 21135360);   // 1536*512
  unsigned short* aggwT  = (unsigned short*)(ws + 22708224);   // 1024*4096
  unsigned short* xH     = (unsigned short*)(ws + 31096832);   // 131072*160
  unsigned short* hH     = (unsigned short*)(ws + 73039872);   // 131072*512
  unsigned short* zqH    = (unsigned short*)(ws + 207257600);  // 131072*256

  // one-time converts (weights also transposed to N x K)
  f32_to_bf16_k<<<(8192 * 1024 + 255) / 256, 256, 0, stream>>>(deter, deterH,
                                                               8192 * 1024);
  f32_to_bf16_T_k<<<dim3(1024 / 32, 2048 / 32), 256, 0, stream>>>(proj_w, projwT, 1024, 2048);
  f32_to_bf16_T_k<<<dim3(160 / 32, 512 / 32),   256, 0, stream>>>(mlp_w1, w1T, 160, 512);
  f32_to_bf16_T_k<<<dim3(512 / 32, 1536 / 32),  256, 0, stream>>>(mlp_w2, w2T, 512, 1536);
  f32_to_bf16_T_k<<<dim3(4096 / 32, 1024 / 32), 256, 0, stream>>>(agg_w, aggwT, 4096, 1024);
  fill_posemb_k<<<(131072 * 32) / 256, 256, 0, stream>>>(pos_emb, xH);

  // proj: (8192,1024)x(1024,2048) -> x (scatter + concat layout)
  gemm_bf16_wmma<0><<<dim3(2048 / BN, 8192 / BM), 256, 0, stream>>>(
      deterH, 1024, projwT, proj_b, nullptr, xH, 1024);

  // mlp1: (131072,160)x(160,512) -> SiLU -> h
  gemm_bf16_wmma<1><<<dim3(512 / BN, 131072 / BM), 256, 0, stream>>>(
      xH, 160, w1T, mlp_b1, nullptr, hH, 160);

  // mlp2: (131072,512)x(512,1536) -> logits (fp32 NT, level-split into d_out)
  float* logits = out + 8388608;                    // after stoch region
  gemm_bf16_wmma<2><<<dim3(1536 / BN, 131072 / BM), 256, 0, stream>>>(
      hH, 512, w2T, mlp_b2, logits, nullptr, 512);

  // argmax + multi-level gather -> z_q (bf16), indices (as float)
  float* idxOut = out + 8388608 + 3 * 67108864;     // after logits region
  argmax_gather_k<<<131072 / 8, 256, 0, stream>>>(logits, idxOut,
                                                  emb_l0, emb_l1, emb_l2, zqH);

  // agg: (8192,4096)x(4096,1024) -> stoch (fp32 at d_out base)
  gemm_bf16_wmma<3><<<dim3(1024 / BN, 8192 / BM), 256, 0, stream>>>(
      zqH, 4096, aggwT, agg_b, out, nullptr, 4096);
}
